// GatedTorsion_90915867721743
// MI455X (gfx1250) — matrix operand order repaired and checked by reading, not compiled, not run on hardware
//
#include <hip/hip_runtime.h>
#include <hip/hip_bf16.h>

// ---------------------------------------------------------------------------
// GatedTorsion fully-fused bf16-WMMA pipeline for gfx1250 (MI455X).
//   - ~2.05 TFLOP total, gather source + weights L2-resident -> matrix-bound.
//   - bf16 WMMA 16x16x32 (f32 accum); weights pre-packed into B fragments.
//   - M=32 rows/WG, 512 threads = 16 wave32 waves; each wave: 2 n-tiles x
//     2 M-tiles.  K-loops are manually 2x unrolled and fenced with
//     sched_barrier(0) to cap live fragments (8) and kill scratch spills.
//   - Gather uses global_load_async_to_lds_b128 (ASYNCcnt) - CDNA5 async path.
//   - LDS: 64K x-tile + 32K act + 64K sym(f32) = 160KB -> 2 WGs per WGP.
// ---------------------------------------------------------------------------

typedef __bf16 bf16_t;
typedef bf16_t v16bf __attribute__((ext_vector_type(16)));
typedef float  v8f   __attribute__((ext_vector_type(8)));

#define F_REP 256
#define F_BET 512
#define M_ROWS 32
#define NT_FULL 32          // 512 / 16 n-tiles
#define NTHREADS 512
#define MAGNITUDE 0.001f

#define WMMA_BF16(a, b, c) \
  __builtin_amdgcn_wmma_f32_16x16x32_bf16(false, (a), false, (b), (short)0, (c), false, false)

#if __has_builtin(__builtin_amdgcn_sched_barrier)
#define SCHED_FENCE() __builtin_amdgcn_sched_barrier(0)
#else
#define SCHED_FENCE() asm volatile("" ::: "memory")
#endif

__device__ __forceinline__ float eluf(float x) {
  return x > 0.f ? x : (__expf(x) - 1.f);
}

union FragU { v16bf v; uint4 u[2]; };

// B fragment from pre-packed weights: tile = kt*NT + nt, 32 lanes x 16 bf16 contiguous.
__device__ __forceinline__ v16bf load_b_pk(const bf16_t* wpk, int tile, int lane) {
  const bf16_t* p = wpk + ((size_t)tile * 32 + lane) * 16;
  FragU f;
  f.u[0] = *(const uint4*)(p);
  f.u[1] = *(const uint4*)(p + 8);
  return f.v;
}

// --------------------------- preprocessing kernels -------------------------

__global__ void cvt_h_kernel(const float* __restrict__ h, bf16_t* __restrict__ out, int n) {
  int i = blockIdx.x * blockDim.x + threadIdx.x;
  if (i < n) out[i] = (bf16_t)h[i];
}

// Repack row-major f32 W[K][N] into per-lane WMMA B fragments (bf16), N padded to Npad.
// packed[((kt*NT + nt)*32 + lane)*16 + j] = W[kt*32 + (lane<16?0:8) + j + (j<8?0:8)][nt*16 + (lane&15)]
__global__ void pack_w_kernel(const float* __restrict__ W, bf16_t* __restrict__ out,
                              int K, int N, int Npad) {
  int idx = blockIdx.x * blockDim.x + threadIdx.x;
  int total = (K / 32) * (Npad / 16) * 512;
  if (idx >= total) return;
  int j    = idx & 15;
  int lane = (idx >> 4) & 31;
  int tile = idx >> 9;
  int NT   = Npad / 16;
  int kt   = tile / NT;
  int nt   = tile % NT;
  int k = kt * 32 + ((lane < 16) ? 0 : 8) + j + ((j < 8) ? 0 : 8);
  int n = nt * 16 + (lane & 15);
  float v = (n < N) ? W[(size_t)k * N + n] : 0.f;
  out[idx] = (bf16_t)v;
}

// ------------------------------- main kernel -------------------------------

__global__ __launch_bounds__(NTHREADS) void torsion_kernel(
    const long long* __restrict__ idxs,
    const bf16_t* __restrict__ hbf,
    const bf16_t* __restrict__ pWs0, const bf16_t* __restrict__ pWs1,
    const bf16_t* __restrict__ pWs2, const bf16_t* __restrict__ pWs3,
    const bf16_t* __restrict__ pWt0, const bf16_t* __restrict__ pWt1,
    const bf16_t* __restrict__ pWt2,
    const bf16_t* __restrict__ pWc0, const bf16_t* __restrict__ pWc1,
    const float* __restrict__ bs0, const float* __restrict__ bs1,
    const float* __restrict__ bs2, const float* __restrict__ bs3,
    const float* __restrict__ bt0, const float* __restrict__ bt1,
    const float* __restrict__ bt2,
    const float* __restrict__ bc0, const float* __restrict__ bc1,
    float* __restrict__ out, int n_tors)
{
  __shared__ bf16_t xbuf[M_ROWS * 1024];   // 64KB: gathered features; later sym(bf16) in lower half
  __shared__ bf16_t hbuf[M_ROWS * 512];    // 32KB: current activation (bf16 A operand)
  __shared__ float  symf[M_ROWS * 512];    // 64KB: stream-0 state, then sym (f32 residual base)
  float* ost = (float*)(xbuf + M_ROWS * 512);  // alias xbuf upper half: 2 x 32 x 16 f32 staging

  const int tid  = threadIdx.x;
  const int lane = tid & 31;
  const int wave = tid >> 5;               // 0..15, owns n-tiles {2*wave, 2*wave+1}
  const int row0 = blockIdx.x * M_ROWS;
  const int nt0  = wave * 2;

  const int colbase = nt0 * 16 + (lane & 15);       // column of n-tile t is colbase + t*16
  const int rowhalf = (lane < 16) ? 0 : 8;          // C-layout row = m*16 + r + rowhalf

  // ---- gather 32 rows x (4 x 256) bf16 features into xbuf via async global->LDS ----
  {
#pragma unroll
    for (int i = 0; i < 8; ++i) {
      int c    = tid + NTHREADS * i;  // 16-byte chunk id, 4096 total
      int r    = c >> 7;              // row 0..31 (128 chunks per row)
      int slot = (c >> 5) & 3;        // feature slot 0..3
      int cw   = c & 31;              // chunk within slot
      int grow = row0 + r;
      if (grow >= n_tors) grow = n_tors - 1;
      long long atom = idxs[(size_t)grow * 4 + slot];
      const bf16_t* src = hbf + (size_t)atom * F_REP + cw * 8;
      bf16_t*       dst = xbuf + r * 1024 + slot * F_REP + cw * 8;
      unsigned long long ga = (unsigned long long)(uintptr_t)src;
      unsigned int       la = (unsigned int)(uintptr_t)dst;
      asm volatile("global_load_async_to_lds_b128 %0, %1, off"
                   :: "v"(la), "v"(ga) : "memory");
    }
#if __has_builtin(__builtin_amdgcn_s_wait_asynccnt)
    __builtin_amdgcn_s_wait_asynccnt(0);
#else
    asm volatile("s_wait_asynccnt 0" ::: "memory");
#endif
  }
  __syncthreads();

  const v8f vzero = {0.f, 0.f, 0.f, 0.f, 0.f, 0.f, 0.f, 0.f};

  // A fragment (16x32 bf16) from LDS row-major tile; mrow selects M-tile (0 or 16).
  auto load_a = [&](const bf16_t* base, int stride, int kbase, int mrow) -> v16bf {
    const int row = (lane & 15) + mrow;
    const int off = (lane < 16) ? 0 : 8;
    const bf16_t* p = base + row * stride + kbase + off;
    FragU f;
    f.u[0] = *(const uint4*)(p);        // K = kbase+off    .. +7
    f.u[1] = *(const uint4*)(p + 16);   // K = kbase+off+16 .. +23
    return f.v;
  };

  // 32x512 @ 512x32(slice) GEMM: 2 n-tiles x 2 M-tiles per wave, B reused 2x.
  // 2x unrolled, sched-fenced: at most 8 fragments live -> no scratch spills.
  auto gemm512 = [&](const bf16_t* A, const bf16_t* W, v8f* acc) {
    for (int kt = 0; kt < 16; kt += 2) {
      v16bf a0 = load_a(A, 512, kt * 32, 0);
      v16bf a1 = load_a(A, 512, kt * 32, 16);
      v16bf b0 = load_b_pk(W, kt * NT_FULL + nt0, lane);
      v16bf b1 = load_b_pk(W, kt * NT_FULL + nt0 + 1, lane);
      v16bf c0 = load_a(A, 512, (kt + 1) * 32, 0);
      v16bf c1 = load_a(A, 512, (kt + 1) * 32, 16);
      v16bf d0 = load_b_pk(W, (kt + 1) * NT_FULL + nt0, lane);
      v16bf d1 = load_b_pk(W, (kt + 1) * NT_FULL + nt0 + 1, lane);
      acc[0] = WMMA_BF16(a0, b0, acc[0]);
      acc[2] = WMMA_BF16(a1, b0, acc[2]);
      acc[1] = WMMA_BF16(a0, b1, acc[1]);
      acc[3] = WMMA_BF16(a1, b1, acc[3]);
      acc[0] = WMMA_BF16(c0, d0, acc[0]);
      acc[2] = WMMA_BF16(c1, d0, acc[2]);
      acc[1] = WMMA_BF16(c0, d1, acc[1]);
      acc[3] = WMMA_BF16(c1, d1, acc[3]);
      SCHED_FENCE();
    }
  };

  auto store_bf16 = [&](bf16_t* dst, const v8f* h) {
#pragma unroll
    for (int m = 0; m < 2; ++m)
#pragma unroll
      for (int t = 0; t < 2; ++t) {
        int col = colbase + t * 16;
#pragma unroll
        for (int r = 0; r < 8; ++r)
          dst[(m * 16 + r + rowhalf) * 512 + col] = (bf16_t)h[m * 2 + t][r];
      }
  };

  // ---------------- symmetrizer: forward + reversed streams ----------------
  const bf16_t* WsL[3] = {pWs1, pWs2, pWs3};
  const float*  bsL[3] = {bs1, bs2, bs3};

  for (int stream = 0; stream < 2; ++stream) {
    // layer 0: 1024 -> 512. Reversed stream remaps K-tiles group-reversed.
    v8f acc[4] = {vzero, vzero, vzero, vzero};
    for (int kt = 0; kt < 32; kt += 2) {
      int ak0 = (stream == 0) ? kt : ((3 - (kt >> 3)) * 8 + (kt & 7));
      int ak1 = (stream == 0) ? (kt + 1) : ((3 - ((kt + 1) >> 3)) * 8 + ((kt + 1) & 7));
      v16bf a0 = load_a(xbuf, 1024, ak0 * 32, 0);
      v16bf a1 = load_a(xbuf, 1024, ak0 * 32, 16);
      v16bf b0 = load_b_pk(pWs0, kt * NT_FULL + nt0, lane);
      v16bf b1 = load_b_pk(pWs0, kt * NT_FULL + nt0 + 1, lane);
      v16bf c0 = load_a(xbuf, 1024, ak1 * 32, 0);
      v16bf c1 = load_a(xbuf, 1024, ak1 * 32, 16);
      v16bf d0 = load_b_pk(pWs0, (kt + 1) * NT_FULL + nt0, lane);
      v16bf d1 = load_b_pk(pWs0, (kt + 1) * NT_FULL + nt0 + 1, lane);
      acc[0] = WMMA_BF16(a0, b0, acc[0]);
      acc[2] = WMMA_BF16(a1, b0, acc[2]);
      acc[1] = WMMA_BF16(a0, b1, acc[1]);
      acc[3] = WMMA_BF16(a1, b1, acc[3]);
      acc[0] = WMMA_BF16(c0, d0, acc[0]);
      acc[2] = WMMA_BF16(c1, d0, acc[2]);
      acc[1] = WMMA_BF16(c0, d1, acc[1]);
      acc[3] = WMMA_BF16(c1, d1, acc[3]);
      SCHED_FENCE();
    }
    v8f hst[4];
#pragma unroll
    for (int m = 0; m < 2; ++m)
#pragma unroll
      for (int t = 0; t < 2; ++t) {
        float bias = bs0[colbase + t * 16];
#pragma unroll
        for (int r = 0; r < 8; ++r) hst[m * 2 + t][r] = eluf(acc[m * 2 + t][r] + bias);
      }

    // residual layers 1..3: h += elu(h @ W + b)
    for (int l = 0; l < 3; ++l) {
      __syncthreads();
      store_bf16(hbuf, hst);
      __syncthreads();
      v8f a2[4] = {vzero, vzero, vzero, vzero};
      gemm512(hbuf, WsL[l], a2);
#pragma unroll
      for (int m = 0; m < 2; ++m)
#pragma unroll
        for (int t = 0; t < 2; ++t) {
          float bias = bsL[l][colbase + t * 16];
#pragma unroll
          for (int r = 0; r < 8; ++r) hst[m * 2 + t][r] += eluf(a2[m * 2 + t][r] + bias);
        }
    }

    if (stream == 0) {
#pragma unroll
      for (int m = 0; m < 2; ++m)
#pragma unroll
        for (int t = 0; t < 2; ++t) {
          int col = colbase + t * 16;
#pragma unroll
          for (int r = 0; r < 8; ++r)
            symf[(m * 16 + r + rowhalf) * 512 + col] = hst[m * 2 + t][r];
        }
    } else {
      // sym = stream0 + stream1; keep f32 in symf, write bf16 copy to xbuf lower half.
      // (all waves finished reading xbuf's layer-0 features many barriers ago)
#pragma unroll
      for (int m = 0; m < 2; ++m)
#pragma unroll
        for (int t = 0; t < 2; ++t) {
          int col = colbase + t * 16;
#pragma unroll
          for (int r = 0; r < 8; ++r) {
            int idx = (m * 16 + r + rowhalf) * 512 + col;
            float s = symf[idx] + hst[m * 2 + t][r];
            symf[idx] = s;
            xbuf[idx] = (bf16_t)s;
          }
        }
    }
  }
  __syncthreads();

  // ---------------- t branch: t = sym + elu(sym@Wt0+bt0); t += elu(t@Wt1+bt1)
  v8f acc[4] = {vzero, vzero, vzero, vzero};
  gemm512(xbuf, pWt0, acc);   // A = sym(bf16) in xbuf lower half, stride 512
  v8f treg[4];
#pragma unroll
  for (int m = 0; m < 2; ++m)
#pragma unroll
    for (int t = 0; t < 2; ++t) {
      int col = colbase + t * 16;
      float bias = bt0[col];
#pragma unroll
      for (int r = 0; r < 8; ++r) {
        int idx = (m * 16 + r + rowhalf) * 512 + col;
        treg[m * 2 + t][r] = symf[idx] + eluf(acc[m * 2 + t][r] + bias);
      }
    }
  __syncthreads();
  store_bf16(hbuf, treg);
  __syncthreads();

  v8f acc2[4] = {vzero, vzero, vzero, vzero};
  gemm512(hbuf, pWt1, acc2);
#pragma unroll
  for (int m = 0; m < 2; ++m)
#pragma unroll
    for (int t = 0; t < 2; ++t) {
      float bias = bt1[colbase + t * 16];
#pragma unroll
      for (int r = 0; r < 8; ++r) treg[m * 2 + t][r] += eluf(acc2[m * 2 + t][r] + bias);
    }
  __syncthreads();
  store_bf16(hbuf, treg);
  __syncthreads();

  // coeffs = t @ Wt2 (512->6, one padded 16-col tile; all waves compute, wave 0 writes)
  {
    v8f a6[2] = {vzero, vzero};
    for (int kt = 0; kt < 16; kt += 2) {
      v16bf a0 = load_a(hbuf, 512, kt * 32, 0);
      v16bf a1 = load_a(hbuf, 512, kt * 32, 16);
      v16bf b  = load_b_pk(pWt2, kt, lane);
      v16bf c0 = load_a(hbuf, 512, (kt + 1) * 32, 0);
      v16bf c1 = load_a(hbuf, 512, (kt + 1) * 32, 16);
      v16bf d  = load_b_pk(pWt2, kt + 1, lane);
      a6[0] = WMMA_BF16(a0, b, a6[0]);
      a6[1] = WMMA_BF16(a1, b, a6[1]);
      a6[0] = WMMA_BF16(c0, d, a6[0]);
      a6[1] = WMMA_BF16(c1, d, a6[1]);
      SCHED_FENCE();
    }
    if (wave == 0) {
      int col = lane & 15;
      float bias = (col < 6) ? bt2[col] : 0.f;
#pragma unroll
      for (int m = 0; m < 2; ++m)
#pragma unroll
        for (int r = 0; r < 8; ++r)
          ost[(m * 16 + r + rowhalf) * 16 + col] = a6[m][r] + bias;
    }
  }

  // ---------------- c branch: c = sym + elu(sym@Wc0+bc0); score = c @ Wc1
  v8f accc[4] = {vzero, vzero, vzero, vzero};
  gemm512(xbuf, pWc0, accc);   // xbuf lower half still holds sym(bf16)
  v8f creg[4];
#pragma unroll
  for (int m = 0; m < 2; ++m)
#pragma unroll
    for (int t = 0; t < 2; ++t) {
      int col = colbase + t * 16;
      float bias = bc0[col];
#pragma unroll
      for (int r = 0; r < 8; ++r) {
        int idx = (m * 16 + r + rowhalf) * 512 + col;
        creg[m * 2 + t][r] = symf[idx] + eluf(accc[m * 2 + t][r] + bias);
      }
    }
  __syncthreads();   // all waves done reading hbuf (coeff GEMM) before overwrite
  store_bf16(hbuf, creg);
  __syncthreads();

  {
    v8f a6[2] = {vzero, vzero};
    for (int kt = 0; kt < 16; kt += 2) {
      v16bf a0 = load_a(hbuf, 512, kt * 32, 0);
      v16bf a1 = load_a(hbuf, 512, kt * 32, 16);
      v16bf b  = load_b_pk(pWc1, kt, lane);
      v16bf c0 = load_a(hbuf, 512, (kt + 1) * 32, 0);
      v16bf c1 = load_a(hbuf, 512, (kt + 1) * 32, 16);
      v16bf d  = load_b_pk(pWc1, kt + 1, lane);
      a6[0] = WMMA_BF16(a0, b, a6[0]);
      a6[1] = WMMA_BF16(a1, b, a6[1]);
      a6[0] = WMMA_BF16(c0, d, a6[0]);
      a6[1] = WMMA_BF16(c1, d, a6[1]);
      SCHED_FENCE();
    }
    if (wave == 0) {
      int col = lane & 15;
      float bias = (col < 6) ? bc1[col] : 0.f;
#pragma unroll
      for (int m = 0; m < 2; ++m)
#pragma unroll
        for (int r = 0; r < 8; ++r)
          ost[32 * 16 + (m * 16 + r + rowhalf) * 16 + col] = a6[m][r] + bias;
    }
  }
  __syncthreads();

  // ---------------- epilogue: out[0]=score, out[1]=coeff*MAG*sigmoid(score)
  if (tid < M_ROWS * 6) {
    int r = tid / 6, j = tid % 6;
    int grow = row0 + r;
    if (grow < n_tors) {
      float coeff = ost[r * 16 + j];
      float score = ost[32 * 16 + r * 16 + j];
      float gate  = 1.f / (1.f + __expf(-score));
      out[(size_t)grow * 6 + j] = score;
      out[(size_t)n_tors * 6 + (size_t)grow * 6 + j] = coeff * MAGNITUDE * gate;
    }
  }
}

// ------------------------------- launcher ----------------------------------

extern "C" void kernel_launch(void* const* d_in, const int* in_sizes, int n_in,
                              void* d_out, int out_size, void* d_ws, size_t ws_size,
                              hipStream_t stream) {
  const float*     h    = (const float*)d_in[0];
  const long long* idxs = (const long long*)d_in[1];
  const float* Ws0 = (const float*)d_in[2];   const float* bs0 = (const float*)d_in[3];
  const float* Ws1 = (const float*)d_in[4];   const float* bs1 = (const float*)d_in[5];
  const float* Ws2 = (const float*)d_in[6];   const float* bs2 = (const float*)d_in[7];
  const float* Ws3 = (const float*)d_in[8];   const float* bs3 = (const float*)d_in[9];
  const float* Wt0 = (const float*)d_in[10];  const float* bt0 = (const float*)d_in[11];
  const float* Wt1 = (const float*)d_in[12];  const float* bt1 = (const float*)d_in[13];
  const float* Wt2 = (const float*)d_in[14];  const float* bt2 = (const float*)d_in[15];
  const float* Wc0 = (const float*)d_in[16];  const float* bc0 = (const float*)d_in[17];
  const float* Wc1 = (const float*)d_in[18];  const float* bc1 = (const float*)d_in[19];

  const int n_atoms = in_sizes[0] / F_REP;
  const int n_tors  = in_sizes[1] / 4;

  // workspace layout (bf16 elements): h_bf16 + packed weights (~30 MB total)
  bf16_t* ws  = (bf16_t*)d_ws;
  size_t  off = 0;
  auto alloc = [&](size_t nelem) { bf16_t* p = ws + off; off += (nelem + 7) & ~(size_t)7; return p; };

  bf16_t* hbf  = alloc((size_t)n_atoms * F_REP);
  bf16_t* pWs0 = alloc((size_t)(1024 / 32) * (512 / 16) * 512);
  bf16_t* pWs1 = alloc((size_t)16 * 32 * 512);
  bf16_t* pWs2 = alloc((size_t)16 * 32 * 512);
  bf16_t* pWs3 = alloc((size_t)16 * 32 * 512);
  bf16_t* pWt0 = alloc((size_t)16 * 32 * 512);
  bf16_t* pWt1 = alloc((size_t)16 * 32 * 512);
  bf16_t* pWt2 = alloc((size_t)16 * 1 * 512);
  bf16_t* pWc0 = alloc((size_t)16 * 32 * 512);
  bf16_t* pWc1 = alloc((size_t)16 * 1 * 512);

  // preprocessing: h -> bf16; weights -> packed bf16 B fragments
  {
    int n = n_atoms * F_REP;
    cvt_h_kernel<<<(n + 255) / 256, 256, 0, stream>>>(h, hbf, n);
  }
  auto pack = [&](const float* W, bf16_t* dst, int K, int N, int Npad) {
    int total = (K / 32) * (Npad / 16) * 512;
    pack_w_kernel<<<(total + 255) / 256, 256, 0, stream>>>(W, dst, K, N, Npad);
  };
  pack(Ws0, pWs0, 1024, 512, 512);
  pack(Ws1, pWs1, 512, 512, 512);
  pack(Ws2, pWs2, 512, 512, 512);
  pack(Ws3, pWs3, 512, 512, 512);
  pack(Wt0, pWt0, 512, 512, 512);
  pack(Wt1, pWt1, 512, 512, 512);
  pack(Wt2, pWt2, 512, 6, 16);
  pack(Wc0, pWc0, 512, 512, 512);
  pack(Wc1, pWc1, 512, 6, 16);

  // fused main kernel: 32 rows / workgroup, 16 wave32 waves
  int blocks = (n_tors + M_ROWS - 1) / M_ROWS;
  torsion_kernel<<<blocks, NTHREADS, 0, stream>>>(
      idxs, hbf,
      pWs0, pWs1, pWs2, pWs3, pWt0, pWt1, pWt2, pWc0, pWc1,
      bs0, bs1, bs2, bs3, bt0, bt1, bt2, bc0, bc1,
      (float*)d_out, n_tors);
}
